// Block_61787399520259
// MI455X (gfx1250) — matrix operand order repaired and checked by reading, not compiled, not run on hardware
//
#include <hip/hip_runtime.h>

typedef unsigned short u16;
typedef __attribute__((ext_vector_type(16))) __bf16 v16bf;
typedef __attribute__((ext_vector_type(8)))  float  v8f;

#define Bn 64
#define Tn 256
#define Cn 384
#define Hn 6
#define HSn 64
#define NROW (Bn * Tn)   // 16384

#if __has_builtin(__builtin_amdgcn_global_load_async_to_lds_b128) && \
    __has_builtin(__builtin_amdgcn_s_wait_asynccnt)
#define HAVE_ASYNC_LDS 1
typedef int v4i __attribute__((vector_size(16)));
typedef __attribute__((address_space(1))) v4i* gas_v4i_p;
typedef __attribute__((address_space(3))) v4i* las_v4i_p;
#else
#define HAVE_ASYNC_LDS 0
#endif

union F16x16 { v16bf v; u16 u[16]; };

__device__ __forceinline__ u16 f2b(float f) {
  unsigned u = __builtin_bit_cast(unsigned, f);
  unsigned r = (u + 0x7FFFu + ((u >> 16) & 1u)) >> 16;
  return (u16)r;
}

// ---------------- weight conversion ----------------
__global__ void cvt_copy(const float* __restrict__ src, u16* __restrict__ dst, int n) {
  int i = blockIdx.x * blockDim.x + threadIdx.x;
  if (i < n) dst[i] = f2b(src[i]);
}

// Wq/Wk/Wv: src is [H][C][HS]; dst is [C][H*HS] (row = c = K dim, col = h*64+d)
__global__ void cvt_qkv(const float* __restrict__ src, u16* __restrict__ dst) {
  int i = blockIdx.x * blockDim.x + threadIdx.x;
  if (i >= Cn * Cn) return;
  int c = i / Cn;
  int j = i % Cn;
  int h = j >> 6, d = j & 63;
  dst[i] = f2b(src[((size_t)h * Cn + c) * HSn + d]);
}

// ---------------- LayerNorm (fp32 in -> bf16 out) ----------------
__global__ __launch_bounds__(128) void ln_k(const float* __restrict__ x,
                                            const float* __restrict__ g,
                                            const float* __restrict__ be,
                                            u16* __restrict__ out) {
  int row = blockIdx.x;
  int tid = threadIdx.x;
  const float* xr = x + (size_t)row * Cn;
  float v0 = xr[tid], v1 = xr[tid + 128], v2 = xr[tid + 256];
  float s1 = v0 + v1 + v2;
  float s2 = v0 * v0 + v1 * v1 + v2 * v2;
  #pragma unroll
  for (int off = 16; off; off >>= 1) {
    s1 += __shfl_xor(s1, off);
    s2 += __shfl_xor(s2, off);
  }
  __shared__ float w1[4], w2[4], stat[2];
  if ((tid & 31) == 0) { w1[tid >> 5] = s1; w2[tid >> 5] = s2; }
  __syncthreads();
  if (tid == 0) {
    float a = w1[0] + w1[1] + w1[2] + w1[3];
    float b = w2[0] + w2[1] + w2[2] + w2[3];
    float mu = a * (1.0f / Cn);
    float var = b * (1.0f / Cn) - mu * mu;
    stat[0] = mu;
    stat[1] = rsqrtf(var + 1e-3f);
  }
  __syncthreads();
  float mu = stat[0], rs = stat[1];
  u16* orow = out + (size_t)row * Cn;
  orow[tid]       = f2b((v0 - mu) * rs * g[tid]       + be[tid]);
  orow[tid + 128] = f2b((v1 - mu) * rs * g[tid + 128] + be[tid + 128]);
  orow[tid + 256] = f2b((v2 - mu) * rs * g[tid + 256] + be[tid + 256]);
}

// ---------------- GEMM epilogue helper ----------------
__device__ __forceinline__ void epi_store(float v, int row, int col, int N,
                                          const float* bias, const float* res,
                                          int relu, int qkv,
                                          u16* outb, float* outf) {
  if (bias) v += bias[col];
  if (res)  v += res[(size_t)row * N + col];
  if (relu) v = fmaxf(v, 0.f);
  if (outf) outf[(size_t)row * N + col] = v;
  if (outb) {
    size_t dst;
    if (qkv) {
      int bb = row >> 8, t = row & 255, hh = col >> 6, d = col & 63;
      dst = (((size_t)(bb * Hn + hh)) * Tn + t) * HSn + d;
    } else {
      dst = (size_t)row * N + col;
    }
    outb[dst] = f2b(v);
  }
}

// ---------------- generic WMMA GEMM, 64x64 tile / 256 threads ----------------
// C[M,N] = A[M,K] (bf16 row-major) * B[K,N] (bf16 row-major)
// epilogue: +bias (f32, per col), +res (f32 [M,N]), relu, out bf16 and/or f32.
// qkv!=0: bf16 output scattered to [b][h][t][d] layout.
// 8 waves in a 4(M) x 2(N) grid; each wave: one A frag, two B frags,
// two v_wmma per K-step (A-fragment reuse).
__global__ __launch_bounds__(256) void gemm_bf16(
    const u16* __restrict__ A, const u16* __restrict__ Bm,
    const float* __restrict__ bias, const float* __restrict__ res,
    u16* __restrict__ outb, float* __restrict__ outf,
    int M, int N, int K, int relu, int qkv) {
  __shared__ u16 As[64][40];   // [m][k]
  __shared__ u16 Bs[64][40];   // transposed: [n][k]

  int tid = threadIdx.x;
  int tileN = blockIdx.x * 64;
  int tileM = blockIdx.y * 64;
  int lane = tid & 31, wave = tid >> 5;
  int mi = wave & 3;        // 0..3 -> 16-row strip
  int ni = wave >> 2;       // 0..1 -> 32-col strip
  int lm = lane & 15;
  int kbase = (lane < 16) ? 0 : 8;
  int koff  = (lane < 16) ? 0 : 16;
  int ra = tid >> 2, ca = (tid & 3) * 8;   // A: 64 rows x 32 k, 16B per thread
  int rb = tid >> 3, cb = (tid & 7) * 8;   // B: 32 k-rows x 64 n, 16B per thread

  v8f acc0 = {0.f, 0.f, 0.f, 0.f, 0.f, 0.f, 0.f, 0.f};
  v8f acc1 = {0.f, 0.f, 0.f, 0.f, 0.f, 0.f, 0.f, 0.f};

  #pragma unroll 2
  for (int k0 = 0; k0 < K; k0 += 32) {
    __syncthreads();
    const u16* ag = A + (size_t)(tileM + ra) * K + k0 + ca;
    const u16* bg = Bm + (size_t)(k0 + rb) * N + tileN + cb;
#if HAVE_ASYNC_LDS
    // contiguous 16B global -> LDS per lane, ASYNCcnt-tracked, bypasses VGPRs
    __builtin_amdgcn_global_load_async_to_lds_b128(
        (gas_v4i_p)(void*)ag, (las_v4i_p)(void*)&As[ra][ca], 0, 0);
#else
    #pragma unroll
    for (int i = 0; i < 8; i++) As[ra][ca + i] = ag[i];
#endif
    #pragma unroll
    for (int i = 0; i < 8; i++) Bs[cb + i][rb] = bg[i];   // transpose into LDS
    if (k0 + 32 < K) {
      __builtin_prefetch(ag + 32, 0, 3);
      __builtin_prefetch(Bm + (size_t)(k0 + 32 + rb) * N + tileN + cb, 0, 3);
    }
#if HAVE_ASYNC_LDS
    __builtin_amdgcn_s_wait_asynccnt(0);
#endif
    __syncthreads();

    F16x16 af, bf0, bf1;
    #pragma unroll
    for (int e = 0; e < 16; e++) {
      int kk = koff + e;
      af.u[e]  = As[mi * 16 + lm][kk];
      bf0.u[e] = Bs[ni * 32 + lm][koff + e];
      bf1.u[e] = Bs[ni * 32 + 16 + lm][koff + e];
    }
    acc0 = __builtin_amdgcn_wmma_f32_16x16x32_bf16(false, af.v, false, bf0.v,
                                                   (short)0, acc0, false, false);
    acc1 = __builtin_amdgcn_wmma_f32_16x16x32_bf16(false, af.v, false, bf1.v,
                                                   (short)0, acc1, false, false);
  }

  int mbase = (lane < 16) ? 0 : 8;
  int col0 = tileN + ni * 32 + lm;
  #pragma unroll
  for (int r = 0; r < 8; r++) {
    int row = tileM + mi * 16 + mbase + r;
    epi_store(acc0[r], row, col0,      N, bias, res, relu, qkv, outb, outf);
    epi_store(acc1[r], row, col0 + 16, N, bias, res, relu, qkv, outb, outf);
  }
}

// ---------------- fused causal attention ----------------
// grid: (B*H*16) blocks of 32 threads; block bx: bh = bx>>4, qi = bx&15
// q,k,v in [B*H][T][HS] bf16; o written as [B][T][H*HS] bf16 (row-major 16384x384)
__global__ __launch_bounds__(32) void attn_k(const u16* __restrict__ q,
                                             const u16* __restrict__ k,
                                             const u16* __restrict__ v,
                                             u16* __restrict__ o) {
  int bx = blockIdx.x;
  int qi = bx & 15;
  int bh = bx >> 4;
  int b = bh / Hn, h = bh % Hn;
  int t0 = qi * 16;

  __shared__ float sbuf[16][256];
  __shared__ u16 pbuf[16][256];

  int lane = threadIdx.x;
  int lm = lane & 15;
  int kbase = (lane < 16) ? 0 : 8;
  int koff  = (lane < 16) ? 0 : 16;
  int mbase = (lane < 16) ? 0 : 8;

  // zero P buffer (covers K-chunk zero-padding for odd 16-col tails)
  u16* pflat = &pbuf[0][0];
  #pragma unroll
  for (int i = 0; i < 128; i++) pflat[lane * 128 + i] = 0;

  // load Q fragments (16 rows x 64 d, split into two K=32 chunks)
  const u16* qrow = q + ((size_t)bh * Tn + t0 + lm) * HSn;
  F16x16 a0, a1;
  #pragma unroll
  for (int e = 0; e < 16; e++) {
    int kk = koff + e;
    a0.u[e] = qrow[kk];
    a1.u[e] = qrow[32 + kk];
  }

  const float scale = 0.125f; // 1/sqrt(64)

  // S = Q K^T for s-tiles 0..qi
  for (int sj = 0; sj <= qi; sj++) {
    int s0 = sj * 16;
    const u16* krow = k + ((size_t)bh * Tn + s0 + lm) * HSn;
    F16x16 b0, b1;
    #pragma unroll
    for (int e = 0; e < 16; e++) {
      b0.u[e] = krow[koff + e];
      b1.u[e] = krow[32 + koff + e];
    }
    v8f c = {0.f, 0.f, 0.f, 0.f, 0.f, 0.f, 0.f, 0.f};
    c = __builtin_amdgcn_wmma_f32_16x16x32_bf16(false, a0.v, false, b0.v, (short)0, c, false, false);
    c = __builtin_amdgcn_wmma_f32_16x16x32_bf16(false, a1.v, false, b1.v, (short)0, c, false, false);
    #pragma unroll
    for (int r = 0; r < 8; r++) {
      int tloc = mbase + r;
      float vv = c[r] * scale;
      if (s0 + lm > t0 + tloc) vv = -1e30f;   // causal mask (diagonal tile only)
      sbuf[tloc][s0 + lm] = vv;
    }
  }
  asm volatile("s_wait_dscnt 0" ::: "memory");

  // row-wise softmax over computed columns; write P as bf16
  int scount = (qi + 1) * 16;
  if (lane < 16) {
    float mx = -1e30f;
    for (int s = 0; s < scount; s++) mx = fmaxf(mx, sbuf[lane][s]);
    float sum = 0.f;
    for (int s = 0; s < scount; s++) sum += __expf(sbuf[lane][s] - mx);
    float inv = 1.0f / sum;
    for (int s = 0; s < scount; s++)
      pbuf[lane][s] = f2b(__expf(sbuf[lane][s] - mx) * inv);
  }
  asm volatile("s_wait_dscnt 0" ::: "memory");

  // O = P @ V  (16 x 64), K-chunks of 32 over s
  int nch = (scount + 31) >> 5;
  for (int nt = 0; nt < 4; nt++) {
    v8f acc = {0.f, 0.f, 0.f, 0.f, 0.f, 0.f, 0.f, 0.f};
    int d = nt * 16 + lm;
    for (int kc = 0; kc < nch; kc++) {
      F16x16 pa, vb;
      #pragma unroll
      for (int e = 0; e < 16; e++) {
        int s = kc * 32 + koff + e;
        pa.u[e] = pbuf[lm][s];
      }
      #pragma unroll
      for (int e = 0; e < 16; e++) {
        int s = kc * 32 + koff + e;
        vb.u[e] = v[((size_t)bh * Tn + s) * HSn + d];
      }
      acc = __builtin_amdgcn_wmma_f32_16x16x32_bf16(false, pa.v, false, vb.v,
                                                    (short)0, acc, false, false);
    }
    #pragma unroll
    for (int r = 0; r < 8; r++) {
      int t = t0 + mbase + r;
      o[((size_t)b * Tn + t) * Cn + h * HSn + nt * 16 + lm] = f2b(acc[r]);
    }
  }
}

// ---------------- host-side launcher ----------------
extern "C" void kernel_launch(void* const* d_in, const int* in_sizes, int n_in,
                              void* d_out, int out_size, void* d_ws, size_t ws_size,
                              hipStream_t stream) {
  const float* x   = (const float*)d_in[0];
  const float* Wq  = (const float*)d_in[1];
  const float* Wk  = (const float*)d_in[2];
  const float* Wv  = (const float*)d_in[3];
  const float* Wp  = (const float*)d_in[4];
  const float* bp  = (const float*)d_in[5];
  const float* W1  = (const float*)d_in[6];
  const float* b1  = (const float*)d_in[7];
  const float* W2  = (const float*)d_in[8];
  const float* b2  = (const float*)d_in[9];
  const float* g1  = (const float*)d_in[10];
  const float* be1 = (const float*)d_in[11];
  const float* g2  = (const float*)d_in[12];
  const float* be2 = (const float*)d_in[13];

  // workspace layout (u16 element offsets); total ~92 MB
  u16* wsp = (u16*)d_ws;
  const size_t CC = (size_t)Cn * Cn;        // 147456
  const size_t C4 = (size_t)Cn * 4 * Cn;    // 589824
  const size_t ACT = (size_t)NROW * Cn;     // 6291456
  size_t off = 0;
  u16* wq_b = wsp + off; off += CC;
  u16* wk_b = wsp + off; off += CC;
  u16* wv_b = wsp + off; off += CC;
  u16* wp_b = wsp + off; off += CC;
  u16* w1_b = wsp + off; off += C4;
  u16* w2_b = wsp + off; off += C4;
  u16* h_b  = wsp + off; off += ACT;
  u16* qb   = wsp + off; off += ACT;   // qb also base of FFN1 activation (4*ACT region)
  u16* kb   = wsp + off; off += ACT;
  u16* vb   = wsp + off; off += ACT;
  u16* ob   = wsp + off; off += ACT;
  u16* ab   = qb;                       // [16384 x 1536] bf16, reuses q/k/v/o region
  float* x1 = (float*)(wsp + off);      // [16384 x 384] fp32

  // 1. convert weights to bf16
  cvt_copy<<<(int)((CC + 255) / 256), 256, 0, stream>>>(Wp, wp_b, (int)CC);
  cvt_copy<<<(int)((C4 + 255) / 256), 256, 0, stream>>>(W1, w1_b, (int)C4);
  cvt_copy<<<(int)((C4 + 255) / 256), 256, 0, stream>>>(W2, w2_b, (int)C4);
  cvt_qkv<<<(int)((CC + 255) / 256), 256, 0, stream>>>(Wq, wq_b);
  cvt_qkv<<<(int)((CC + 255) / 256), 256, 0, stream>>>(Wk, wk_b);
  cvt_qkv<<<(int)((CC + 255) / 256), 256, 0, stream>>>(Wv, wv_b);

  // 2. LN1
  ln_k<<<NROW, 128, 0, stream>>>(x, g1, be1, h_b);

  // 3. QKV projections (scattered to [b][h][t][d])
  dim3 gqkv(Cn / 64, NROW / 64);
  gemm_bf16<<<gqkv, 256, 0, stream>>>(h_b, wq_b, nullptr, nullptr, qb, nullptr,
                                      NROW, Cn, Cn, 0, 1);
  gemm_bf16<<<gqkv, 256, 0, stream>>>(h_b, wk_b, nullptr, nullptr, kb, nullptr,
                                      NROW, Cn, Cn, 0, 1);
  gemm_bf16<<<gqkv, 256, 0, stream>>>(h_b, wv_b, nullptr, nullptr, vb, nullptr,
                                      NROW, Cn, Cn, 0, 1);

  // 4. fused causal attention -> O [16384 x 384] bf16
  attn_k<<<Bn * Hn * (Tn / 16), 32, 0, stream>>>(qb, kb, vb, ob);

  // 5. output projection + residual -> x1 fp32
  gemm_bf16<<<dim3(Cn / 64, NROW / 64), 256, 0, stream>>>(
      ob, wp_b, bp, x, nullptr, x1, NROW, Cn, Cn, 0, 0);

  // 6. LN2
  ln_k<<<NROW, 128, 0, stream>>>(x1, g2, be2, h_b);

  // 7. FFN1: relu(h2 @ W1 + b1) -> bf16 [16384 x 1536]
  gemm_bf16<<<dim3(4 * Cn / 64, NROW / 64), 256, 0, stream>>>(
      h_b, w1_b, b1, nullptr, ab, nullptr, NROW, 4 * Cn, Cn, 1, 0);

  // 8. FFN2 + residual -> d_out fp32
  gemm_bf16<<<dim3(Cn / 64, NROW / 64), 256, 0, stream>>>(
      ab, w2_b, b2, x1, nullptr, (float*)d_out, NROW, Cn, 4 * Cn, 0, 0);
}
